// AutoformerSpace_67894843015798
// MI455X (gfx1250) — compile-verified, matrix operand order checked by reading, not run on hardware
//
#include <hip/hip_runtime.h>

// ---------------------------------------------------------------------------
// AutoformerSpace attention block for MI455X (gfx1250): wave32 + bf16 WMMA +
// double-buffered async global->LDS copies (ASYNCcnt path).
//   0. cvt_bf16 / cvt_tr_bf16 : x -> bf16; Wq,Wk,Wv,Wproj -> transposed bf16
//   1. setup_maps : rel-pos index maps (u8) + packed value-RPE table (bf16)
//   2. qkv_gemm   : xb @ Wt -> q,k bf16 (B,H,N,64) and v^T bf16 (B,H,64,208)
//   3. pb_kernel  : Pb[bh,n,j] = q . tab_k_{v,h}[j]   (separable score bias)
//   4. attn_kernel: flash attention; S=Q K^T (wmma) + gathered bias, online
//                   softmax, O+=P V (wmma); value-RPE via 30-bin histogram
//   5. proj_gemm  : attn_out @ Wproj^T + bproj -> f32 d_out
// All WMMA operands are staged K-contiguous so fragment loads are ds_load_b128.
// ---------------------------------------------------------------------------

typedef __bf16 bf16;
typedef __attribute__((ext_vector_type(16))) __bf16 bf16x16;
typedef __attribute__((ext_vector_type(8)))  float  f32x8;

#define NHEADS 10
#define HD     64
#define SEQ    197
#define SEQP   208                 // padded seq stride for v^T (416B, 16B mult)
#define BATCH  64
#define EMB    640
#define BHD    (BATCH * NHEADS)    // 640
#define ROWS   (BATCH * SEQ)       // 12608

// ---- CDNA5 async global->LDS (ASYNCcnt) -----------------------------------
__device__ __forceinline__ unsigned lds_addr32(const void* p) {
  return (unsigned)(unsigned long long)p;   // generic LDS ptr low 32b == LDS addr
}
__device__ __forceinline__ void async_copy_b128(unsigned lds_off, const void* g) {
  asm volatile("global_load_async_to_lds_b128 %0, %1, off"
               :: "v"(lds_off), "v"(g) : "memory");
}
__device__ __forceinline__ void wait_async0() {
  asm volatile("s_wait_asynccnt 0x0" ::: "memory");
}
__device__ __forceinline__ void wait_async4() {
  asm volatile("s_wait_asynccnt 0x4" ::: "memory");  // 4 in-flight = next tile
}

// ---- WMMA operand helpers (CDNA5 16-bit 16x16x32 layouts, wave32) ---------
// A (16xK): lane m = lane&15, half = lane>>4; element e -> K = e + 8*half + (e&8)
// B (Kx16): lane n = lane&15, same K map (so a transposed-stored B == load_a).
// C/D: m = r + 8*half, n = lane&15.

__device__ __forceinline__ int kmap(int e, int half) {
  return e + 8 * half + (e & 8);
}

__device__ __forceinline__ f32x8 wmma_bf16(bf16x16 a, bf16x16 b, f32x8 c) {
  return __builtin_amdgcn_wmma_f32_16x16x32_bf16(false, a, false, b, (short)0, c,
                                                 false, false);
}

// operand from lane-row-major tile (row0+idx, kofs+k); K contiguous -> b128
__device__ __forceinline__ bf16x16 load_a(const bf16* p, int row0, int kofs, int ld) {
  const int lane = threadIdx.x & 31, m = lane & 15, half = lane >> 4;
  bf16x16 a;
#pragma unroll
  for (int e = 0; e < 16; ++e) a[e] = p[(row0 + m) * ld + kofs + kmap(e, half)];
  return a;
}

// B operand from K-major tile (used only for the tiny vtab epilogue)
__device__ __forceinline__ bf16x16 load_b(const bf16* p, int kofs, int col0, int ld) {
  const int lane = threadIdx.x & 31, n = lane & 15, half = lane >> 4;
  bf16x16 b;
#pragma unroll
  for (int e = 0; e < 16; ++e) b[e] = p[(kofs + kmap(e, half)) * ld + col0 + n];
  return b;
}

// A operand from 16x16 bf16 tile, K zero-padded to 32
__device__ __forceinline__ bf16x16 load_a_k16(const bf16* p) {
  const int lane = threadIdx.x & 31, m = lane & 15, half = lane >> 4;
  bf16x16 a;
#pragma unroll
  for (int e = 0; e < 16; ++e) {
    int k = kmap(e, half);
    a[e] = (k < 16) ? p[m * 16 + k] : (bf16)0.0f;
  }
  return a;
}

// A operand converted on the fly from f32 source
__device__ __forceinline__ bf16x16 load_a_f32(const float* p, int kofs, int ld) {
  const int lane = threadIdx.x & 31, m = lane & 15, half = lane >> 4;
  bf16x16 a;
#pragma unroll
  for (int e = 0; e < 16; ++e) a[e] = (bf16)p[m * ld + kofs + kmap(e, half)];
  return a;
}

// ---------------------------------------------------------------------------
// 0a. f32 -> bf16            0b. f32 -> bf16 transposed (640x640)
// ---------------------------------------------------------------------------
__global__ void cvt_bf16(const float* __restrict__ src, bf16* __restrict__ dst,
                         int n) {
  int i = blockIdx.x * blockDim.x + threadIdx.x;
  if (i < n) dst[i] = (bf16)src[i];
}

__global__ void cvt_tr_bf16(const float* __restrict__ src, bf16* __restrict__ dst) {
  int idx = blockIdx.x * blockDim.x + threadIdx.x;
  if (idx < EMB * EMB) {
    int c = idx / EMB, k = idx % EMB;
    dst[idx] = (bf16)src[(size_t)k * EMB + c];   // dst[c][k] = src[k][c]
  }
}

// ---------------------------------------------------------------------------
// 1. Relative-position index maps + packed value-RPE table.
// ---------------------------------------------------------------------------
__global__ void setup_maps(unsigned char* __restrict__ ivm,
                           unsigned char* __restrict__ ihm,
                           const float* __restrict__ tvv,
                           const float* __restrict__ tvh,
                           bf16* __restrict__ vtab) {
  int idx = blockIdx.x * blockDim.x + threadIdx.x;
  if (idx < SEQ * SEQ) {
    int qi = idx / SEQ, ki = idx % SEQ;
    int v = 0, h = 0;
    if (qi > 0 && ki > 0) {
      int i = qi - 1, j = ki - 1;
      int dv = j / 14 - i / 14;
      int dh = j % 14 - i % 14;
      dv = dv > 14 ? 14 : (dv < -14 ? -14 : dv);
      dh = dh > 14 ? 14 : (dh < -14 ? -14 : dh);
      v = dv + 15;
      h = dh + 15;
    }
    ivm[idx] = (unsigned char)v;
    ihm[idx] = (unsigned char)h;
  }
  if (idx < 64 * 64) {
    int j = idx >> 6, d = idx & 63;
    float val = 0.0f;
    if (j < 30)                 val = tvv[j * 64 + d];
    else if (j >= 32 && j < 62) val = tvh[(j - 32) * 64 + d];
    vtab[idx] = (bf16)val;
  }
}

// ---------------------------------------------------------------------------
// 2. QKV GEMM, double-buffered async tiles. grid = (30, 197), 128 threads.
//    A tile: xb 64x32.  B tile: Wt rows cbase..cbase+63, 64x32. Both LDT=40.
// ---------------------------------------------------------------------------
__global__ __launch_bounds__(128)
void qkv_gemm(const bf16* __restrict__ xb,
              const bf16* __restrict__ Wqt, const bf16* __restrict__ Wkt,
              const bf16* __restrict__ Wvt,
              bf16* __restrict__ qb, bf16* __restrict__ kb,
              bf16* __restrict__ vt) {
  const int LDT = 40;                       // 80B rows (16B aligned)
  __shared__ bf16 As[2][64 * LDT];
  __shared__ bf16 Bs[2][64 * LDT];

  const int bm  = blockIdx.y * 64;
  const int sel = blockIdx.x / NHEADS;
  const int h   = blockIdx.x % NHEADS;
  const bf16* Wt = (sel == 0) ? Wqt : (sel == 1 ? Wkt : Wvt);
  const float mult = (sel == 0) ? 0.125f : 1.0f;   // hd^-0.5 folded into q
  const int cbase = h * 64;

  const int tid = threadIdx.x;
  const int wave = tid >> 5;
  const int wm = (wave & 1) * 32, wn = (wave >> 1) * 32;

  auto issue_tile = [&](int k0, int buf) {
#pragma unroll
    for (int t = 0; t < 2; ++t) {
      int id = t * 128 + tid;
      int r = id >> 2, cc = id & 3;              // 64 rows x 4 16B chunks
      async_copy_b128(lds_addr32(&As[buf][r * LDT + cc * 8]),
                      &xb[(size_t)(bm + r) * EMB + k0 + cc * 8]);
      async_copy_b128(lds_addr32(&Bs[buf][r * LDT + cc * 8]),
                      &Wt[(size_t)(cbase + r) * EMB + k0 + cc * 8]);
    }
  };

  const f32x8 zf = {0.f, 0.f, 0.f, 0.f, 0.f, 0.f, 0.f, 0.f};
  f32x8 c[2][2];
#pragma unroll
  for (int i = 0; i < 2; ++i)
#pragma unroll
    for (int j = 0; j < 2; ++j) c[i][j] = zf;

  issue_tile(0, 0);
  const int NK = EMB / 32;
  for (int it = 0; it < NK; ++it) {
    const int cur = it & 1;
    if (it + 1 < NK) {
      issue_tile((it + 1) * 32, cur ^ 1);
      wait_async4();                 // in-order: <=4 in flight => tile it landed
    } else {
      wait_async0();
    }
    __syncthreads();
#pragma unroll
    for (int i = 0; i < 2; ++i) {
      bf16x16 a = load_a(&As[cur][0], wm + i * 16, 0, LDT);
#pragma unroll
      for (int j = 0; j < 2; ++j) {
        bf16x16 b = load_a(&Bs[cur][0], wn + j * 16, 0, LDT);
        c[i][j] = wmma_bf16(a, b, c[i][j]);
      }
    }
    __syncthreads();
  }

  // epilogue: uniform div once, conditional wrap per element (m < 64 < SEQ)
  const int bb0 = bm / SEQ;
  const int n0  = bm - bb0 * SEQ;
  const int lane = tid & 31, halfl = lane >> 4, nn = lane & 15;
#pragma unroll
  for (int i = 0; i < 2; ++i)
#pragma unroll
    for (int j = 0; j < 2; ++j)
#pragma unroll
      for (int r = 0; r < 8; ++r) {
        int m = wm + i * 16 + r + 8 * halfl;
        int d = wn + j * 16 + nn;
        int n = n0 + m, bb = bb0;
        if (n >= SEQ) { n -= SEQ; ++bb; }
        size_t bh = (size_t)(bb * NHEADS + h);
        bf16 val = (bf16)(c[i][j][r] * mult);
        if (sel == 2) vt[(bh * HD + d) * SEQP + n] = val;   // v^T (bh, d, seq)
        else if (sel == 0) qb[(bh * SEQ + n) * HD + d] = val;
        else               kb[(bh * SEQ + n) * HD + d] = val;
      }
}

// ---------------------------------------------------------------------------
// 3. Pb[bh*SEQ+n][col]: col 0..29 = q.tab_k_v[col], 32..61 = q.tab_k_h.
// ---------------------------------------------------------------------------
__global__ __launch_bounds__(64)
void pb_kernel(const bf16* __restrict__ qb, const float* __restrict__ tkv,
               const float* __restrict__ tkh, float* __restrict__ Pb) {
  __shared__ float qrow[64];
  const int row = blockIdx.x;
  const int t = threadIdx.x;
  qrow[t] = (float)qb[(size_t)row * HD + t];
  __syncthreads();
  float out = 0.0f;
  const float* tab = nullptr;
  if (t < 30)                 tab = tkv + t * HD;
  else if (t >= 32 && t < 62) tab = tkh + (t - 32) * HD;
  if (tab) {
#pragma unroll 8
    for (int d = 0; d < HD; ++d) out += qrow[d] * tab[d];
  }
  Pb[(size_t)row * 64 + t] = out;
}

// ---------------------------------------------------------------------------
// 4. Fused flash attention. grid = (13, 640), 32 threads (1 wave).
//    V staged from v^T so PV fragment loads are K-contiguous (b128).
// ---------------------------------------------------------------------------
__global__ __launch_bounds__(32)
void attn_kernel(const bf16* __restrict__ qb, const bf16* __restrict__ kb,
                 const bf16* __restrict__ vt, const float* __restrict__ Pb,
                 const unsigned char* __restrict__ ivm,
                 const unsigned char* __restrict__ ihm,
                 const bf16* __restrict__ vtab, bf16* __restrict__ aout) {
  const int LDS_T = 72;                          // K tile rows: 144B
  __shared__ bf16  Qs[16 * LDS_T];
  __shared__ bf16  Ks[16 * LDS_T];
  __shared__ bf16  Vts[64 * 32];                 // v^T tile: 64 d-rows x 32 keys
  __shared__ bf16  Ps[16 * 16];
  __shared__ float bins[16 * 64];
  __shared__ float fcorr[16];

  const int bh = blockIdx.y;
  const int q0 = blockIdx.x * 16;
  const int lane = threadIdx.x;
  const int half = lane >> 4, nn = lane & 15;

  const bf16* qrow = qb + (size_t)bh * SEQ * HD;
  const bf16* krow = kb + (size_t)bh * SEQ * HD;
  const bf16* vrow = vt + (size_t)bh * HD * SEQP;

  for (int i = lane; i < 16 * 64; i += 32) {
    int r = i >> 6, d = i & 63;
    int qg = q0 + r;
    Qs[r * LDS_T + d] = (qg < SEQ) ? qrow[(size_t)qg * HD + d] : (bf16)0.0f;
  }
  for (int i = lane; i < 16 * 64; i += 32) bins[i] = 0.0f;
  // zero upper key-half of Vts once; async tiles only write keys 0..15
  for (int i = lane; i < 64 * 16; i += 32)
    Vts[(i >> 4) * 32 + 16 + (i & 15)] = (bf16)0.0f;
  __syncthreads();

  const bf16x16 aq0 = load_a(Qs, 0, 0, LDS_T);
  const bf16x16 aq1 = load_a(Qs, 0, 32, LDS_T);
  const f32x8 zf = {0.f, 0.f, 0.f, 0.f, 0.f, 0.f, 0.f, 0.f};
  f32x8 acc[4];
#pragma unroll
  for (int t = 0; t < 4; ++t) acc[t] = zf;
  float rowmax[8], rowsum[8];
#pragma unroll
  for (int r = 0; r < 8; ++r) { rowmax[r] = -3.0e4f; rowsum[r] = 0.0f; }

  for (int kt = 0; kt < 13; ++kt) {
    const int k0 = kt * 16;
    __syncthreads();
    if (kt < 12) {
#pragma unroll
      for (int t = 0; t < 4; ++t) {
        int id = t * 32 + lane;
        int r = id >> 3, cc = id & 7;            // K: 16 rows x 8 chunks
        async_copy_b128(lds_addr32(&Ks[r * LDS_T + cc * 8]),
                        &krow[(size_t)(k0 + r) * HD + cc * 8]);
        int d = id >> 1, vc = id & 1;            // V^T: 64 rows x 2 chunks
        async_copy_b128(lds_addr32(&Vts[d * 32 + vc * 8]),
                        &vrow[(size_t)d * SEQP + k0 + vc * 8]);
      }
      wait_async0();
    } else {
      // partial last tile: guarded sync path zero-fills masked keys
      for (int i = lane; i < 16 * 64; i += 32) {
        int r = i >> 6, d = i & 63;
        int kg = k0 + r;
        Ks[r * LDS_T + d] = (kg < SEQ) ? krow[(size_t)kg * HD + d] : (bf16)0.0f;
      }
      for (int i = lane; i < 64 * 16; i += 32) {
        int d = i >> 4, kl = i & 15;
        int kg = k0 + kl;
        Vts[d * 32 + kl] = (kg < SEQ) ? vrow[(size_t)d * SEQP + kg] : (bf16)0.0f;
      }
    }
    __syncthreads();

    // S = Q K^T : K^T as B operand == A-style load of row-major K tile
    f32x8 s = wmma_bf16(aq0, load_a(Ks, 0, 0, LDS_T), zf);
    s = wmma_bf16(aq1, load_a(Ks, 0, 32, LDS_T), s);

    // bias gather + mask
    int jv[8], jh[8];
    float p[8];
#pragma unroll
    for (int r = 0; r < 8; ++r) {
      int m = r + 8 * half;
      int qg = q0 + m, kg = k0 + nn;
      float sv = s[r];
      if (qg < SEQ && kg < SEQ) {
        int jvv = ivm[qg * SEQ + kg];
        int jhh = ihm[qg * SEQ + kg];
        jv[r] = jvv; jh[r] = jhh;
        const float* pb = Pb + ((size_t)bh * SEQ + qg) * 64;
        sv += pb[jvv] + pb[32 + jhh];
      } else {
        sv = -3.0e4f; jv[r] = -1; jh[r] = 0;
      }
      s[r] = sv;
    }

    // online softmax (row reductions across 16-lane groups)
#pragma unroll
    for (int r = 0; r < 8; ++r) {
      float tmax = s[r];
      tmax = fmaxf(tmax, __shfl_xor(tmax, 1, 32));
      tmax = fmaxf(tmax, __shfl_xor(tmax, 2, 32));
      tmax = fmaxf(tmax, __shfl_xor(tmax, 4, 32));
      tmax = fmaxf(tmax, __shfl_xor(tmax, 8, 32));
      float nm = fmaxf(rowmax[r], tmax);
      float corr = __expf(rowmax[r] - nm);
      rowmax[r] = nm;
      float pv = __expf(s[r] - nm);
      p[r] = pv;
      float ts = pv;
      ts += __shfl_xor(ts, 1, 32);
      ts += __shfl_xor(ts, 2, 32);
      ts += __shfl_xor(ts, 4, 32);
      ts += __shfl_xor(ts, 8, 32);
      rowsum[r] = rowsum[r] * corr + ts;
#pragma unroll
      for (int t = 0; t < 4; ++t) acc[t][r] *= corr;
      if (nn == 0) fcorr[r + 8 * half] = corr;
    }
    __syncthreads();
    for (int i = lane; i < 16 * 64; i += 32) bins[i] *= fcorr[i >> 6];
    __syncthreads();
#pragma unroll
    for (int r = 0; r < 8; ++r) {
      int m = r + 8 * half;
      Ps[m * 16 + nn] = (bf16)p[r];
      if (jv[r] >= 0) {
        atomicAdd(&bins[m * 64 + jv[r]], p[r]);
        atomicAdd(&bins[m * 64 + 32 + jh[r]], p[r]);
      }
    }
    __syncthreads();

    // O += P V  (V^T rows are d, K-contiguous; keys 16..31 are zero)
    const bf16x16 pa = load_a_k16(Ps);
#pragma unroll
    for (int t = 0; t < 4; ++t)
      acc[t] = wmma_bf16(pa, load_a(Vts, t * 16, 0, 32), acc[t]);
  }

  __syncthreads();
  // value-RPE: acc += bins(16x64) x vtab(64x64)
  const bf16x16 ba0 = load_a_f32(bins, 0, 64);
  const bf16x16 ba1 = load_a_f32(bins, 32, 64);
#pragma unroll
  for (int t = 0; t < 4; ++t) {
    acc[t] = wmma_bf16(ba0, load_b(vtab, 0, t * 16, 64), acc[t]);
    acc[t] = wmma_bf16(ba1, load_b(vtab, 32, t * 16, 64), acc[t]);
  }

  const int b = bh / NHEADS, h = bh % NHEADS;
#pragma unroll
  for (int r = 0; r < 8; ++r) {
    int m = r + 8 * half;
    int qg = q0 + m;
    if (qg < SEQ) {
      float inv = 1.0f / rowsum[r];
#pragma unroll
      for (int t = 0; t < 4; ++t) {
        int d = t * 16 + nn;
        aout[((size_t)(b * SEQ + qg)) * EMB + h * HD + d] =
            (bf16)(acc[t][r] * inv);
      }
    }
  }
}

// ---------------------------------------------------------------------------
// 5. Projection GEMM (+bias) -> f32 out. grid = (10, 197), 128 threads.
// ---------------------------------------------------------------------------
__global__ __launch_bounds__(128)
void proj_gemm(const bf16* __restrict__ A, const bf16* __restrict__ Wpt,
               const float* __restrict__ bias, float* __restrict__ out) {
  const int LDT = 40;
  __shared__ bf16 As[2][64 * LDT];
  __shared__ bf16 Bs[2][64 * LDT];

  const int bm = blockIdx.y * 64, bn = blockIdx.x * 64;
  const int tid = threadIdx.x;
  const int wave = tid >> 5;
  const int wm = (wave & 1) * 32, wn = (wave >> 1) * 32;

  auto issue_tile = [&](int k0, int buf) {
#pragma unroll
    for (int t = 0; t < 2; ++t) {
      int id = t * 128 + tid;
      int r = id >> 2, cc = id & 3;
      async_copy_b128(lds_addr32(&As[buf][r * LDT + cc * 8]),
                      &A[(size_t)(bm + r) * EMB + k0 + cc * 8]);
      async_copy_b128(lds_addr32(&Bs[buf][r * LDT + cc * 8]),
                      &Wpt[(size_t)(bn + r) * EMB + k0 + cc * 8]);
    }
  };

  const f32x8 zf = {0.f, 0.f, 0.f, 0.f, 0.f, 0.f, 0.f, 0.f};
  f32x8 c[2][2];
#pragma unroll
  for (int i = 0; i < 2; ++i)
#pragma unroll
    for (int j = 0; j < 2; ++j) c[i][j] = zf;

  issue_tile(0, 0);
  const int NK = EMB / 32;
  for (int it = 0; it < NK; ++it) {
    const int cur = it & 1;
    if (it + 1 < NK) {
      issue_tile((it + 1) * 32, cur ^ 1);
      wait_async4();
    } else {
      wait_async0();
    }
    __syncthreads();
#pragma unroll
    for (int i = 0; i < 2; ++i) {
      bf16x16 a = load_a(&As[cur][0], wm + i * 16, 0, LDT);
#pragma unroll
      for (int j = 0; j < 2; ++j) {
        bf16x16 b = load_a(&Bs[cur][0], wn + j * 16, 0, LDT);
        c[i][j] = wmma_bf16(a, b, c[i][j]);
      }
    }
    __syncthreads();
  }

  const int lane = tid & 31, halfl = lane >> 4, nn = lane & 15;
#pragma unroll
  for (int i = 0; i < 2; ++i)
#pragma unroll
    for (int j = 0; j < 2; ++j)
#pragma unroll
      for (int r = 0; r < 8; ++r) {
        int row = bm + wm + i * 16 + r + 8 * halfl;
        int col = bn + wn + j * 16 + nn;
        out[(size_t)row * EMB + col] = c[i][j][r] + bias[col];
      }
}

// ---------------------------------------------------------------------------
extern "C" void kernel_launch(void* const* d_in, const int* in_sizes, int n_in,
                              void* d_out, int out_size, void* d_ws, size_t ws_size,
                              hipStream_t stream) {
  (void)in_sizes; (void)n_in; (void)out_size; (void)ws_size;

  const float* x     = (const float*)d_in[0];
  const float* Wq    = (const float*)d_in[1];
  const float* Wk    = (const float*)d_in[2];
  const float* Wv    = (const float*)d_in[3];
  const float* Wproj = (const float*)d_in[4];
  const float* bproj = (const float*)d_in[5];
  const float* tkv   = (const float*)d_in[6];
  const float* tkh   = (const float*)d_in[7];
  const float* tvv   = (const float*)d_in[8];
  const float* tvh   = (const float*)d_in[9];
  float* out = (float*)d_out;

  char* ws = (char*)d_ws;
  size_t off = 0;
  auto alloc = [&](size_t bytes) -> void* {
    void* p = ws + off;
    off += (bytes + 255) & ~(size_t)255;
    return p;
  };

  bf16*  xb   = (bf16*)alloc((size_t)ROWS * EMB * sizeof(bf16));        // 16.1 MB
  bf16*  wqt  = (bf16*)alloc((size_t)EMB * EMB * sizeof(bf16));
  bf16*  wkt  = (bf16*)alloc((size_t)EMB * EMB * sizeof(bf16));
  bf16*  wvt  = (bf16*)alloc((size_t)EMB * EMB * sizeof(bf16));
  bf16*  wpt  = (bf16*)alloc((size_t)EMB * EMB * sizeof(bf16));
  bf16*  qb   = (bf16*)alloc((size_t)BHD * SEQ * HD * sizeof(bf16));    // 16.1 MB
  bf16*  kbuf = (bf16*)alloc((size_t)BHD * SEQ * HD * sizeof(bf16));    // 16.1 MB
  bf16*  vt   = (bf16*)alloc((size_t)BHD * HD * SEQP * sizeof(bf16));   // 17.0 MB
  float* Pb   = (float*)alloc((size_t)BHD * SEQ * 64 * sizeof(float));  // 32.3 MB
  bf16*  aout = (bf16*)alloc((size_t)ROWS * EMB * sizeof(bf16));        // 16.1 MB
  unsigned char* ivm = (unsigned char*)alloc(SEQ * SEQ);
  unsigned char* ihm = (unsigned char*)alloc(SEQ * SEQ);
  bf16*  vtab = (bf16*)alloc(64 * 64 * sizeof(bf16));

  const int nx = ROWS * EMB, nw = EMB * EMB;
  cvt_bf16<<<dim3((nx + 255) / 256), 256, 0, stream>>>(x, xb, nx);
  cvt_tr_bf16<<<dim3((nw + 255) / 256), 256, 0, stream>>>(Wq, wqt);
  cvt_tr_bf16<<<dim3((nw + 255) / 256), 256, 0, stream>>>(Wk, wkt);
  cvt_tr_bf16<<<dim3((nw + 255) / 256), 256, 0, stream>>>(Wv, wvt);
  cvt_tr_bf16<<<dim3((nw + 255) / 256), 256, 0, stream>>>(Wproj, wpt);
  setup_maps<<<dim3((SEQ * SEQ + 255) / 256), 256, 0, stream>>>(
      ivm, ihm, tvv, tvh, vtab);
  qkv_gemm<<<dim3(3 * NHEADS, ROWS / 64), 128, 0, stream>>>(
      xb, wqt, wkt, wvt, qb, kbuf, vt);
  pb_kernel<<<dim3(BHD * SEQ), 64, 0, stream>>>(qb, tkv, tkh, Pb);
  attn_kernel<<<dim3((SEQ + 15) / 16, BHD), 32, 0, stream>>>(
      qb, kbuf, vt, Pb, ivm, ihm, vtab, aout);
  proj_gemm<<<dim3(EMB / 64, ROWS / 64), 128, 0, stream>>>(
      aout, wpt, bproj, out);
}
